// RelativeMultiHeadAttention_2019_Dai_5351529250950
// MI455X (gfx1250) — matrix operand order repaired and checked by reading, not compiled, not run on hardware
//
#include <hip/hip_runtime.h>

typedef __attribute__((ext_vector_type(16))) __bf16 v16bf;
typedef __attribute__((ext_vector_type(8)))  float  v8f;

#define S_LEN   2048
#define D_MODEL 1024
#define NHEAD   16
#define DHEAD   64

__device__ __forceinline__ __bf16 f2bf(float x) {
  union { float f; unsigned u; } v; v.f = x;
  unsigned r = v.u + 0x7FFFu + ((v.u >> 16) & 1u);   // round-to-nearest-even
  unsigned short h = (unsigned short)(r >> 16);
  __bf16 o; __builtin_memcpy(&o, &h, 2); return o;
}

__device__ __forceinline__ v8f zero_v8f() {
  v8f z;
#pragma unroll
  for (int i = 0; i < 8; ++i) z[i] = 0.0f;
  return z;
}

__device__ __forceinline__ v8f wmma_bf16(v16bf a, v16bf b, v8f c) {
  return __builtin_amdgcn_wmma_f32_16x16x32_bf16(false, a, false, b, (short)0, c, false, false);
}

// A fragment: 16x32 (MxK). Lane L: row = L&15, half h = L>>4 holds K = h*8+[0..7] and 16+h*8+[0..7].
__device__ __forceinline__ v16bf load_a(const __bf16* base, int row0, int maxrow, int ld, int kofs) {
  int lane = threadIdx.x & 31;
  int r = lane & 15, h = lane >> 4;
  int row = row0 + r; if (row > maxrow) row = maxrow;
  const __bf16* p0 = base + (size_t)row * ld + kofs + h * 8;
  v16bf a;
#pragma unroll
  for (int t = 0; t < 8; ++t) { a[t] = p0[t]; a[8 + t] = p0[16 + t]; }
  return a;
}

// B fragment: 32x16 (KxN). Lane L: col = L&15, half h = L>>4 holds K = h*8+[0..7] and 16+h*8+[0..7].
// Column c's K-data is contiguous at base + c*ld.
__device__ __forceinline__ v16bf load_b(const __bf16* base, int col0, int maxcol, int ld, int kofs) {
  int lane = threadIdx.x & 31;
  int c = lane & 15, h = lane >> 4;
  int col = col0 + c; if (col > maxcol) col = maxcol;
  const __bf16* p0 = base + (size_t)col * ld + kofs + h * 8;
  v16bf b;
#pragma unroll
  for (int t = 0; t < 8; ++t) { b[t] = p0[t]; b[8 + t] = p0[16 + t]; }
  return b;
}

// per-lane 16B async DMA global -> LDS (tracked by ASYNCcnt, no VGPR data path)
__device__ __forceinline__ void async_b128(unsigned lds_addr, unsigned long long gaddr) {
  asm volatile("global_load_async_to_lds_b128 %0, %1, off"
               :: "v"(lds_addr), "v"(gaddr) : "memory");
}
__device__ __forceinline__ void wait_async0() {
  asm volatile("s_wait_asynccnt 0x0" ::: "memory");
}
__device__ __forceinline__ unsigned lds_off(const void* p) {
  return (unsigned)(unsigned long long)p;   // generic->LDS: low 32 bits are the LDS offset
}

// ---------------- fp32 -> bf16 convert ----------------
__global__ __launch_bounds__(256)
void k_convert(const float* __restrict__ src, __bf16* __restrict__ dst, int n) {
  int i = blockIdx.x * blockDim.x + threadIdx.x;
  if (i < n) dst[i] = f2bf(src[i]);
}

// ---------------- LDS-staged GEMM: Y = X @ W^T + bias ----------------
// X: [M,K] bf16 row-major, W: [N,K] bf16 row-major.
// mode 0: Y bf16 [B,H,S,64];  mode 1: Y bf16 [B,H,64,S] (V transposed);  mode 2: Y fp32 [M,N].
// Block: 256 threads = 8 waves; block tile 128(M) x 64(N); k-slice 32, double-buffered async->LDS.
__global__ __launch_bounds__(256)
void k_gemm(const __bf16* __restrict__ X, const __bf16* __restrict__ W,
            const float* __restrict__ bias, __bf16* __restrict__ Ybf,
            float* __restrict__ Yf, int M, int N, int K, int mode) {
  __shared__ __bf16 sA[2][128 * 32];
  __shared__ __bf16 sB[2][64 * 32];

  int ntn = N >> 6;
  int mtb = blockIdx.x / ntn, ntb = blockIdx.x - mtb * ntn;
  int m0 = mtb << 7, n0 = ntb << 6;
  if (m0 >= M) return;
  int tid = threadIdx.x, wv = tid >> 5;
  int mw = (wv >> 1) * 32;          // wave row offset inside block tile
  int nw = (wv & 1) * 32;           // wave col offset inside block tile

  unsigned laddrA[2] = { lds_off(&sA[0][0]), lds_off(&sA[1][0]) };
  unsigned laddrB[2] = { lds_off(&sB[0][0]), lds_off(&sB[1][0]) };

  // chunk mapping: A tile 128x32 bf16 = 512 x 16B (2 chunks/thread); B tile 64x32 = 256 x 16B (1/thread)
  int arow0 = tid >> 1, ahalf = tid & 1;           // chunks tid and tid+256 -> rows tid>>1, 128+?? no:
  // chunk ch = tid + 256*r: row = ch>>1, half = ch&1  (r=0: rows 0..127 even halves pattern)
  int bcol = tid >> 2, bq = tid & 3;

  // issue k-slice 0
  {
#pragma unroll
    for (int r = 0; r < 2; ++r) {
      int ch = tid + 256 * r;
      int row = ch >> 1, half = ch & 1;
      async_b128(laddrA[0] + (unsigned)(row * 32 + half * 16) * 2,
                 (unsigned long long)(X + (size_t)(m0 + row) * K + half * 16));
    }
    async_b128(laddrB[0] + (unsigned)(bcol * 32 + bq * 8) * 2,
               (unsigned long long)(W + (size_t)(n0 + bcol) * K + bq * 8));
  }
  (void)arow0; (void)ahalf;

  v8f a00 = zero_v8f(), a01 = zero_v8f(), a10 = zero_v8f(), a11 = zero_v8f();
  int NT = K >> 5;
  for (int kt = 0; kt < NT; ++kt) {
    wait_async0();
    __syncthreads();
    if (kt + 1 < NT) {
      int nb = (kt + 1) & 1, k0 = (kt + 1) << 5;
#pragma unroll
      for (int r = 0; r < 2; ++r) {
        int ch = tid + 256 * r;
        int row = ch >> 1, half = ch & 1;
        async_b128(laddrA[nb] + (unsigned)(row * 32 + half * 16) * 2,
                   (unsigned long long)(X + (size_t)(m0 + row) * K + k0 + half * 16));
      }
      async_b128(laddrB[nb] + (unsigned)(bcol * 32 + bq * 8) * 2,
                 (unsigned long long)(W + (size_t)(n0 + bcol) * K + k0 + bq * 8));
    }
    const __bf16* a = &sA[kt & 1][0];
    const __bf16* b = &sB[kt & 1][0];
    v16bf fa0 = load_a(a, mw,      127, 32, 0);
    v16bf fa1 = load_a(a, mw + 16, 127, 32, 0);
    v16bf fb0 = load_b(b, nw,       63, 32, 0);
    v16bf fb1 = load_b(b, nw + 16,  63, 32, 0);
    a00 = wmma_bf16(fa0, fb0, a00);
    a01 = wmma_bf16(fa0, fb1, a01);
    a10 = wmma_bf16(fa1, fb0, a10);
    a11 = wmma_bf16(fa1, fb1, a11);
  }

  int lane = tid & 31, col = lane & 15, half = lane >> 4;
#pragma unroll
  for (int mi = 0; mi < 2; ++mi)
#pragma unroll
    for (int ni = 0; ni < 2; ++ni) {
      v8f acc = mi ? (ni ? a11 : a10) : (ni ? a01 : a00);
      int n = n0 + nw + ni * 16 + col;
      float bv = bias[n];
      int hh = n >> 6, d = n & 63;
#pragma unroll
      for (int i = 0; i < 8; ++i) {
        int m = m0 + mw + mi * 16 + i + 8 * half;
        float val = acc[i] + bv;
        if (mode == 2) {
          Yf[(size_t)m * N + n] = val;
        } else {
          int b = m >> 11, s = m & (S_LEN - 1);
          size_t idx = (mode == 1)
            ? (((size_t)b * NHEAD + hh) * DHEAD + d) * S_LEN + s
            : (((size_t)b * NHEAD + hh) * S_LEN + s) * DHEAD + d;
          Ybf[idx] = f2bf(val);
        }
      }
    }
}

// ---------------- fused relative attention (flash-style, one wave = 16 query rows) --------------
// Q,K,P: [B*H, S, 64] bf16.  Vt: [B*H, 64, S] bf16.  Ctx: [B, S, 1024] bf16.
__global__ __launch_bounds__(256)
void k_rel_attn(const __bf16* __restrict__ Q, const __bf16* __restrict__ K,
                const __bf16* __restrict__ Vt, const __bf16* __restrict__ P,
                __bf16* __restrict__ Ctx) {
  __shared__ float  s_w1[8][16][48];
  __shared__ float  s_w2[8][16][48];
  __shared__ __bf16 s_pr[8][16][32];

  int wv  = threadIdx.x >> 5;
  int wid = blockIdx.x * 8 + wv;
  int qt  = wid & (S_LEN / 16 - 1);       // 128 query tiles
  int bh  = wid >> 7;                     // 0..31 = b*H + h
  if (bh >= 32) return;
  int i0 = qt * 16;

  const __bf16* q  = Q  + (size_t)bh * S_LEN * DHEAD;
  const __bf16* k  = K  + (size_t)bh * S_LEN * DHEAD;
  const __bf16* p  = P  + (size_t)bh * S_LEN * DHEAD;
  const __bf16* vt = Vt + (size_t)bh * DHEAD * S_LEN;

  int lane = threadIdx.x & 31, col = lane & 15, half = lane >> 4;

  v16bf qa0  = load_a(q, i0,     S_LEN - 1, DHEAD, 0);
  v16bf qa1  = load_a(q, i0,     S_LEN - 1, DHEAD, 32);
  v16bf qsa0 = load_a(q, i0 + 1, S_LEN - 1, DHEAD, 0);
  v16bf qsa1 = load_a(q, i0 + 1, S_LEN - 1, DHEAD, 32);

  float mrow[8], lrow[8];
#pragma unroll
  for (int i = 0; i < 8; ++i) { mrow[i] = -3.0e38f; lrow[i] = 0.0f; }
  v8f o0 = zero_v8f(), o1 = zero_v8f(), o2 = zero_v8f(), o3 = zero_v8f();
  const float inv_sqrt_d = 0.03125f;   // 1/sqrt(1024)

  for (int j0 = 0; j0 < S_LEN; j0 += 32) {
    // ---- content scores ----
    v8f s0 = zero_v8f(), s1 = zero_v8f();
    {
      v16bf kb0a = load_b(k, j0,      S_LEN - 1, DHEAD, 0);
      v16bf kb0b = load_b(k, j0,      S_LEN - 1, DHEAD, 32);
      s0 = wmma_bf16(qa0, kb0a, s0);
      s0 = wmma_bf16(qa1, kb0b, s0);
      v16bf kb1a = load_b(k, j0 + 16, S_LEN - 1, DHEAD, 0);
      v16bf kb1b = load_b(k, j0 + 16, S_LEN - 1, DHEAD, 32);
      s1 = wmma_bf16(qa0, kb1a, s1);
      s1 = wmma_bf16(qa1, kb1b, s1);
    }

    // ---- shifted positional scores: two windowed Q@P^T GEMMs + LDS gather ----
    int t_lo = j0 - i0 - 15, t_hi = j0 + 31 - i0;
    bool br1 = (t_lo <= 0), br2 = (t_hi >= 2);
    int c1b = S_LEN - 16 + (j0 - i0);     // = S-1 + t_lo
    int c2b = t_lo - 2; if (c2b < 0) c2b = 0;
    if (br1) {
#pragma unroll
      for (int cb = 0; cb < 3; ++cb) {
        v8f wt = zero_v8f();
        v16bf pb0 = load_b(p, c1b + cb * 16, S_LEN - 1, DHEAD, 0);
        v16bf pb1 = load_b(p, c1b + cb * 16, S_LEN - 1, DHEAD, 32);
        wt = wmma_bf16(qa0, pb0, wt);
        wt = wmma_bf16(qa1, pb1, wt);
#pragma unroll
        for (int i = 0; i < 8; ++i) s_w1[wv][i + 8 * half][cb * 16 + col] = wt[i];
      }
    }
    if (br2) {
#pragma unroll
      for (int cb = 0; cb < 3; ++cb) {
        v8f wt = zero_v8f();
        v16bf pb0 = load_b(p, c2b + cb * 16, S_LEN - 1, DHEAD, 0);
        v16bf pb1 = load_b(p, c2b + cb * 16, S_LEN - 1, DHEAD, 32);
        wt = wmma_bf16(qsa0, pb0, wt);
        wt = wmma_bf16(qsa1, pb1, wt);
#pragma unroll
        for (int i = 0; i < 8; ++i) s_w2[wv][i + 8 * half][cb * 16 + col] = wt[i];
      }
    }
    asm volatile("s_wait_dscnt 0x0" ::: "memory");  // wave-local LDS RAW

    float sc0[8], sc1[8];
#pragma unroll
    for (int i = 0; i < 8; ++i) {
      int r  = i + 8 * half;
      int gi = i0 + r;
      {
        int gj = j0 + col;
        int t = gj - gi;
        float pos = (t == 1) ? 0.0f
                  : (t <= 0) ? s_w1[wv][r][t - t_lo]
                             : s_w2[wv][r][(t - 2) - c2b];
        sc0[i] = (s0[i] + pos) * inv_sqrt_d;
      }
      {
        int gj = j0 + 16 + col;
        int t = gj - gi;
        float pos = (t == 1) ? 0.0f
                  : (t <= 0) ? s_w1[wv][r][t - t_lo]
                             : s_w2[wv][r][(t - 2) - c2b];
        sc1[i] = (s1[i] + pos) * inv_sqrt_d;
      }
    }

    // ---- online softmax ----
#pragma unroll
    for (int i = 0; i < 8; ++i) {
      float mx = fmaxf(sc0[i], sc1[i]);
      mx = fmaxf(mx, __shfl_xor(mx, 1));
      mx = fmaxf(mx, __shfl_xor(mx, 2));
      mx = fmaxf(mx, __shfl_xor(mx, 4));
      mx = fmaxf(mx, __shfl_xor(mx, 8));
      float mnew = fmaxf(mrow[i], mx);
      float resc = __expf(mrow[i] - mnew);
      float e0 = __expf(sc0[i] - mnew);
      float e1 = __expf(sc1[i] - mnew);
      float sum = e0 + e1;
      sum += __shfl_xor(sum, 1);
      sum += __shfl_xor(sum, 2);
      sum += __shfl_xor(sum, 4);
      sum += __shfl_xor(sum, 8);
      lrow[i] = lrow[i] * resc + sum;
      mrow[i] = mnew;
      o0[i] *= resc; o1[i] *= resc; o2[i] *= resc; o3[i] *= resc;
      sc0[i] = e0; sc1[i] = e1;
    }

    // ---- probs -> LDS, then P @ V via WMMA ----
#pragma unroll
    for (int i = 0; i < 8; ++i) {
      s_pr[wv][i + 8 * half][col]      = f2bf(sc0[i]);
      s_pr[wv][i + 8 * half][16 + col] = f2bf(sc1[i]);
    }
    asm volatile("s_wait_dscnt 0x0" ::: "memory");
    v16bf pa = load_a((const __bf16*)&s_pr[wv][0][0], 0, 15, 32, 0);
    {
      v16bf vb0 = load_b(vt + j0, 0,  DHEAD - 1, S_LEN, 0);
      o0 = wmma_bf16(pa, vb0, o0);
      v16bf vb1 = load_b(vt + j0, 16, DHEAD - 1, S_LEN, 0);
      o1 = wmma_bf16(pa, vb1, o1);
      v16bf vb2 = load_b(vt + j0, 32, DHEAD - 1, S_LEN, 0);
      o2 = wmma_bf16(pa, vb2, o2);
      v16bf vb3 = load_b(vt + j0, 48, DHEAD - 1, S_LEN, 0);
      o3 = wmma_bf16(pa, vb3, o3);
    }
  }

  int b = bh >> 4, hh = bh & 15;
#pragma unroll
  for (int i = 0; i < 8; ++i) {
    int gi = i0 + i + 8 * half;
    float inv_l = 1.0f / lrow[i];
    size_t base = ((size_t)b * S_LEN + gi) * D_MODEL + hh * DHEAD;
    Ctx[base +  0 + col] = f2bf(o0[i] * inv_l);
    Ctx[base + 16 + col] = f2bf(o1[i] * inv_l);
    Ctx[base + 32 + col] = f2bf(o2[i] * inv_l);
    Ctx[base + 48 + col] = f2bf(o3[i] * inv_l);
  }
}

// ------------------------------------------------------------------
extern "C" void kernel_launch(void* const* d_in, const int* in_sizes, int n_in,
                              void* d_out, int out_size, void* d_ws, size_t ws_size,
                              hipStream_t stream) {
  (void)in_sizes; (void)n_in; (void)out_size; (void)ws_size;
  const float* query = (const float*)d_in[0];
  const float* key_  = (const float*)d_in[1];
  const float* value = (const float*)d_in[2];
  const float* pos   = (const float*)d_in[3];
  const float* Wq = (const float*)d_in[4];  const float* bq = (const float*)d_in[5];
  const float* Wk = (const float*)d_in[6];  const float* bk = (const float*)d_in[7];
  const float* Wv = (const float*)d_in[8];  const float* bv = (const float*)d_in[9];
  const float* Wp = (const float*)d_in[10]; const float* bp = (const float*)d_in[11];
  const float* Wo = (const float*)d_in[12]; const float* bo = (const float*)d_in[13];
  float* out = (float*)d_out;

  const size_t NX = (size_t)2 * S_LEN * D_MODEL;     // 4,194,304
  const size_t NW = (size_t)D_MODEL * D_MODEL;       // 1,048,576
  __bf16* w = (__bf16*)d_ws;
  size_t off = 0;
  auto carve = [&](size_t n) { __bf16* pp = w + off; off += n; return pp; };
  __bf16 *Xq = carve(NX), *Xk = carve(NX), *Xv = carve(NX), *Xp = carve(NX);
  __bf16 *Bq = carve(NW), *Bk = carve(NW), *Bv = carve(NW), *Bp = carve(NW), *Bo = carve(NW);
  __bf16 *Qh = carve(NX), *Kh = carve(NX), *Ph = carve(NX), *Vt = carve(NX);
  __bf16 *Ctx = carve(NX);

  dim3 blk(256);
  k_convert<<<(unsigned)((NX + 255) / 256), blk, 0, stream>>>(query, Xq, (int)NX);
  k_convert<<<(unsigned)((NX + 255) / 256), blk, 0, stream>>>(key_,  Xk, (int)NX);
  k_convert<<<(unsigned)((NX + 255) / 256), blk, 0, stream>>>(value, Xv, (int)NX);
  k_convert<<<(unsigned)((NX + 255) / 256), blk, 0, stream>>>(pos,   Xp, (int)NX);
  k_convert<<<(unsigned)((NW + 255) / 256), blk, 0, stream>>>(Wq, Bq, (int)NW);
  k_convert<<<(unsigned)((NW + 255) / 256), blk, 0, stream>>>(Wk, Bk, (int)NW);
  k_convert<<<(unsigned)((NW + 255) / 256), blk, 0, stream>>>(Wv, Bv, (int)NW);
  k_convert<<<(unsigned)((NW + 255) / 256), blk, 0, stream>>>(Wp, Bp, (int)NW);
  k_convert<<<(unsigned)((NW + 255) / 256), blk, 0, stream>>>(Wo, Bo, (int)NW);

  // GEMMs: block tile 128x64 -> (4096/128)*(1024/64) = 512 blocks
  const int gemm_blocks = (4096 / 128) * (1024 / 64);
  k_gemm<<<gemm_blocks, blk, 0, stream>>>(Xq, Bq, bq, Qh, nullptr, 4096, 1024, 1024, 0);
  k_gemm<<<gemm_blocks, blk, 0, stream>>>(Xk, Bk, bk, Kh, nullptr, 4096, 1024, 1024, 0);
  k_gemm<<<gemm_blocks, blk, 0, stream>>>(Xv, Bv, bv, Vt, nullptr, 4096, 1024, 1024, 1);
  k_gemm<<<gemm_blocks, blk, 0, stream>>>(Xp, Bp, bp, Ph, nullptr, 4096, 1024, 1024, 0);

  // fused relative attention: 32 (b,h) * 128 q-tiles = 4096 waves = 512 blocks
  k_rel_attn<<<512, blk, 0, stream>>>(Qh, Kh, Vt, Ph, Ctx);

  // output projection -> fp32 d_out
  k_gemm<<<gemm_blocks, blk, 0, stream>>>(Ctx, Bo, bo, nullptr, out, 4096, 1024, 1024, 2);
}
